// CNN_align_81355270520887
// MI455X (gfx1250) — compile-verified, matrix operand order checked by reading, not run on hardware
//
#include <hip/hip_runtime.h>
#include <hip/hip_bf16.h>
#include <math.h>

typedef __attribute__((ext_vector_type(16))) _Float16 v16h;
typedef __attribute__((ext_vector_type(8)))  _Float16 v8h;
typedef __attribute__((ext_vector_type(8)))  float    v8f;

#define DEVFN __device__ __forceinline__

// ---------------------------------------------------------------------------
// CDNA5 WMMA 16-bit A/B fragment layout: lane l, half hi = l>>4;
// element i in v16h maps to K = (i/8)*16 + hi*8 + (i%8)  -> two contiguous
// 8-half (16-byte) runs per lane. So a fragment from K-contiguous memory is
// just two b128 loads concatenated.
// ---------------------------------------------------------------------------
DEVFN v16h concat8(v8h a, v8h b) {
  return __builtin_shufflevector(a, b, 0, 1, 2, 3, 4, 5, 6, 7,
                                 8, 9, 10, 11, 12, 13, 14, 15);
}

// Scalar-gather fallback fragment builder (used only for conv1, K=27).
template <typename F>
DEVFN v16h gather_frag(int lane, F elem) {
  v16h out;
  const int hi = lane >> 4;
#pragma unroll
  for (int i = 0; i < 16; ++i) {
    const int k = ((i >> 3) << 4) + (hi << 3) + (i & 7);
    out[i] = elem(k);
  }
  return out;
}

// ---------------------------------------------------------------------------
// Generic implicit-GEMM conv (NHWC activations, weights TRANSPOSED to [CO][K]),
// relu, f16 out. One wave32 per 16x16 (pixels x channels) tile.
// Fast path (CI % 32 == 0): every 32-wide K chunk sits inside one (r,s) tap,
// so A and B fragments are vector loads. Top-left pad = 0 for all convs here
// (JAX SAME with in = 2*out pads only bottom/right; VALID pads nothing).
// ---------------------------------------------------------------------------
template <int CI, int CO, int HIN, int WIN, int HOUT, int WOUT, int R, int STRIDE, typename TIN>
__global__ void conv_relu_wmma(const TIN* __restrict__ x,
                               const _Float16* __restrict__ wT,  // [CO][K]
                               _Float16* __restrict__ y) {
  constexpr int K = R * R * CI;
  constexpr int M = HOUT * WOUT;
  const int lane = threadIdx.x;
  const int hi = lane >> 4;
  const int m = blockIdx.x * 16 + (lane & 15);
  const int n = blockIdx.y * 16 + (lane & 15);
  const int b = blockIdx.z;
  const int oh = m / WOUT, ow = m % WOUT;
  const bool mval = (m < M);
  const TIN* xb = x + (size_t)b * HIN * WIN * CI;
  const _Float16* wrow = wT + (size_t)n * K + hi * 8;
  v8f acc = {};

  if constexpr ((CI % 32) == 0) {
    for (int k0 = 0; k0 < K; k0 += 32) {
      const int rs = k0 / CI, ci0 = k0 % CI;
      const int ih = oh * STRIDE + rs / R;
      const int iw = ow * STRIDE + rs % R;
      v16h af = {};
      if (mval && ih < HIN && iw < WIN) {
        const _Float16* pa =
            (const _Float16*)xb + (((size_t)ih * WIN + iw) * CI + ci0 + hi * 8);
        af = concat8(*(const v8h*)pa, *(const v8h*)(pa + 16));
      }
      const _Float16* pb = wrow + k0;
      const v16h bf = concat8(*(const v8h*)pb, *(const v8h*)(pb + 16));
      acc = __builtin_amdgcn_wmma_f32_16x16x32_f16(false, af, false, bf,
                                                   (short)0, acc, false, false);
    }
  } else {
    // conv1 path: K = 27 < 32, masked scalar gather, single WMMA.
    for (int k0 = 0; k0 < K; k0 += 32) {
      v16h af = gather_frag(lane, [&](int kk) -> _Float16 {
        const int k = k0 + kk;
        if (!mval || k >= K) return (_Float16)0.0f;
        const int ci = k % CI, rs = k / CI;
        const int ih = oh * STRIDE + rs / R;
        const int iw = ow * STRIDE + rs % R;
        if (ih >= HIN || iw >= WIN) return (_Float16)0.0f;
        return (_Float16)(float)xb[((size_t)ih * WIN + iw) * CI + ci];
      });
      v16h bf = gather_frag(lane, [&](int kk) -> _Float16 {
        const int k = k0 + kk;
        if (k >= K) return (_Float16)0.0f;
        return wT[(size_t)n * K + k];
      });
      acc = __builtin_amdgcn_wmma_f32_16x16x32_f16(false, af, false, bf,
                                                   (short)0, acc, false, false);
    }
  }
#pragma unroll
  for (int v = 0; v < 8; ++v) {
    const int mm = blockIdx.x * 16 + v + (hi << 3);
    if (mm < M) {
      float val = acc[v];
      val = val < 0.0f ? 0.0f : val;
      y[((size_t)b * M + mm) * CO + n] = (_Float16)val;
    }
  }
}

// ---------------------------------------------------------------------------
// Correlation: per batch GEMM, M=N=256 (pixels), K=512 (channels).
// corr[b,m,n] = fA[b,m,:] . fB[b,n,:]. Both operands channel-contiguous ->
// pure b128 fragment loads. Writes f32 (final reduce) + f16 (Wr1 input).
// ---------------------------------------------------------------------------
__global__ void corr_wmma(const _Float16* __restrict__ fA,
                          const _Float16* __restrict__ fB,
                          float* __restrict__ corr,
                          _Float16* __restrict__ corr_h) {
  const int lane = threadIdx.x;
  const int hi = lane >> 4;
  const int b = blockIdx.z;
  const int m = blockIdx.x * 16 + (lane & 15);
  const int n = blockIdx.y * 16 + (lane & 15);
  const _Float16* arow = fA + ((size_t)b * 256 + m) * 512 + hi * 8;
  const _Float16* brow = fB + ((size_t)b * 256 + n) * 512 + hi * 8;
  v8f acc = {};
  for (int k0 = 0; k0 < 512; k0 += 32) {
    __builtin_prefetch(arow + k0 + 256, 0, 1);  // global_prefetch_b8
    const v16h af = concat8(*(const v8h*)(arow + k0), *(const v8h*)(arow + k0 + 16));
    const v16h bf = concat8(*(const v8h*)(brow + k0), *(const v8h*)(brow + k0 + 16));
    acc = __builtin_amdgcn_wmma_f32_16x16x32_f16(false, af, false, bf,
                                                 (short)0, acc, false, false);
  }
#pragma unroll
  for (int v = 0; v < 8; ++v) {
    const int mm = blockIdx.x * 16 + v + (hi << 3);
    const size_t idx = ((size_t)b * 256 + mm) * 256 + n;
    corr[idx] = acc[v];
    corr_h[idx] = (_Float16)acc[v];
  }
}

// ---------------------------------------------------------------------------
// Weight convert + transpose: HWIO f32 [K][CO] -> f16 [CO][K]
// ---------------------------------------------------------------------------
__global__ void cvt_transpose(const float* __restrict__ src,
                              _Float16* __restrict__ dst, int K, int CO) {
  const int i = blockIdx.x * blockDim.x + threadIdx.x;
  if (i >= K * CO) return;
  const int co = i / K, k = i % K;
  dst[i] = (_Float16)src[k * CO + co];
}

// ---------------------------------------------------------------------------
// Channel-wise L2 normalize: one wave per 512-channel pixel vector.
// ---------------------------------------------------------------------------
__global__ void l2norm_512(const _Float16* __restrict__ in,
                           _Float16* __restrict__ out, int npix) {
  const int wid = (blockIdx.x * blockDim.x + threadIdx.x) >> 5;
  const int lane = threadIdx.x & 31;
  if (wid >= npix) return;
  const _Float16* p = in + (size_t)wid * 512;
  float ss = 0.0f;
  for (int c = lane; c < 512; c += 32) {
    const float v = (float)p[c];
    ss += v * v;
  }
#pragma unroll
  for (int off = 16; off > 0; off >>= 1) ss += __shfl_xor(ss, off, 32);
  const float inv = 1.0f / (sqrtf(ss) + 1e-6f);
  _Float16* q = out + (size_t)wid * 512;
  for (int c = lane; c < 512; c += 32) q[c] = (_Float16)((float)p[c] * inv);
}

// ---------------------------------------------------------------------------
// Dense: geo[b] = flatten(r2[b]) @ Wd + bd   (32 x 2304 @ 2304 x 18)
// ---------------------------------------------------------------------------
__global__ void dense_geo(const _Float16* __restrict__ r2,
                          const float* __restrict__ Wd,
                          const float* __restrict__ bd,
                          float* __restrict__ geo) {
  const int b = blockIdx.x;
  const int o = threadIdx.x;
  if (o >= 18) return;
  float s = bd[o];
  const _Float16* rb = r2 + (size_t)b * 2304;
  for (int k = 0; k < 2304; ++k) s += (float)rb[k] * Wd[k * 18 + o];
  geo[b * 18 + o] = s;
}

// ---------------------------------------------------------------------------
// TPS fit: per batch, Gauss-Jordan 12x14 (two RHS at once).
// Output per b (40 floats): dst[9][2], theta_x[1..11], theta_y[1..11].
// ---------------------------------------------------------------------------
__device__ __constant__ float c_src[18] = {
    0.f, 0.f, 0.5f, 0.f, 1.f, 0.f,
    0.f, 0.5f, 0.5f, 0.5f, 1.f, 0.5f,
    0.f, 1.f, 5.f, 1.f, 1.f, 1.f};

DEVFN float tps_u(float r) { return r * r * __logf(r + 1e-6f); }
DEVFN float pdistf(float ax, float ay, float bx, float by) {
  const float dx = ax - bx, dy = ay - by;
  return sqrtf(dx * dx + dy * dy + 1e-12f);
}

__global__ void tps_solve(const float* __restrict__ geo, float* __restrict__ tps) {
  const int b = blockIdx.x * blockDim.x + threadIdx.x;
  if (b >= 32) return;
  float dst[9][2];
#pragma unroll
  for (int i = 0; i < 9; ++i) {
    dst[i][0] = c_src[2 * i + 0] + geo[b * 18 + 2 * i + 0];
    dst[i][1] = c_src[2 * i + 1] + geo[b * 18 + 2 * i + 1];
  }
  float M[12][14];
  for (int i = 0; i < 12; ++i)
    for (int j = 0; j < 14; ++j) M[i][j] = 0.0f;
  for (int i = 0; i < 9; ++i) {
    for (int j = 0; j < 9; ++j)
      M[i][j] = tps_u(pdistf(dst[i][0], dst[i][1], dst[j][0], dst[j][1]));
    M[i][9] = 1.0f;  M[i][10] = dst[i][0];  M[i][11] = dst[i][1];
    M[9][i] = 1.0f;  M[10][i] = dst[i][0];  M[11][i] = dst[i][1];
    M[i][12] = -geo[b * 18 + 2 * i + 0];  // delta = SRC - dst = -mv
    M[i][13] = -geo[b * 18 + 2 * i + 1];
  }
  for (int col = 0; col < 12; ++col) {
    int piv = col;
    float best = fabsf(M[col][col]);
    for (int r = col + 1; r < 12; ++r) {
      const float a = fabsf(M[r][col]);
      if (a > best) { best = a; piv = r; }
    }
    if (piv != col)
      for (int j = col; j < 14; ++j) {
        const float t = M[col][j]; M[col][j] = M[piv][j]; M[piv][j] = t;
      }
    const float d = M[col][col];
    const float inv = (fabsf(d) > 1e-20f) ? 1.0f / d : 0.0f;
    for (int j = col; j < 14; ++j) M[col][j] *= inv;
    for (int r = 0; r < 12; ++r)
      if (r != col) {
        const float f = M[r][col];
        for (int j = col; j < 14; ++j) M[r][j] -= f * M[col][j];
      }
  }
  float* o = tps + b * 40;
  for (int i = 0; i < 9; ++i) { o[2 * i] = dst[i][0]; o[2 * i + 1] = dst[i][1]; }
  for (int i = 0; i < 11; ++i) { o[18 + i] = M[i + 1][12]; o[29 + i] = M[i + 1][13]; }
}

// ---------------------------------------------------------------------------
// Fused inlier mask + reduction: out[b] = sum(corr * mask). Thread t handles
// cell (k,l) = (t/16, t%16); sums corr[b,i,j,k,l] over i,j passing the mask.
// ---------------------------------------------------------------------------
__global__ void mask_reduce(const float* __restrict__ corr,
                            const float* __restrict__ tps,
                            float* __restrict__ out) {
  __shared__ float sh[40];
  __shared__ float red[256];
  const int b = blockIdx.x, t = threadIdx.x;
  if (t < 40) sh[t] = tps[b * 40 + t];
  __syncthreads();
  const int kk = t >> 4, ll = t & 15;
  const float x = ll * (1.0f / 15.0f), y = kk * (1.0f / 15.0f);
  float sx = 0.0f, sy = 0.0f;
#pragma unroll
  for (int i = 0; i < 8; ++i) { sx += sh[18 + i]; sy += sh[29 + i]; }
  float zx = sh[26] + x * sh[27] + y * sh[28];
  float zy = sh[37] + x * sh[38] + y * sh[39];
#pragma unroll
  for (int i = 0; i < 9; ++i) {
    const float u = tps_u(pdistf(x, y, sh[2 * i], sh[2 * i + 1]));
    const float wxi = (i == 0) ? -sx : sh[18 + i - 1];
    const float wyi = (i == 0) ? -sy : sh[29 + i - 1];
    zx += wxi * u;
    zy += wyi * u;
  }
  const float ax = (x + zx) * 15.0f;
  const float ay = (y + zy) * 15.0f;
  float s = 0.0f;
  const float* cb = corr + (size_t)b * 65536;
  for (int i = 0; i < 16; ++i) {
    if (fabsf((float)i - ay) <= 1.0f) {
      for (int j = 0; j < 16; ++j) {
        if (fabsf((float)j - ax) <= 1.0f)
          s += cb[(size_t)(i * 16 + j) * 256 + t];
      }
    }
  }
  red[t] = s;
  __syncthreads();
  for (int stride = 128; stride > 0; stride >>= 1) {
    if (t < stride) red[t] += red[t + stride];
    __syncthreads();
  }
  if (t == 0) out[b] = red[0];
}

// ---------------------------------------------------------------------------
// Host-side orchestration
// ---------------------------------------------------------------------------
extern "C" void kernel_launch(void* const* d_in, const int* in_sizes, int n_in,
                              void* d_out, int out_size, void* d_ws, size_t ws_size,
                              hipStream_t stream) {
  (void)in_sizes; (void)n_in; (void)out_size; (void)ws_size;
  const float* imgA = (const float*)d_in[0];
  const float* imgB = (const float*)d_in[1];
  const float* W1 = (const float*)d_in[2];
  const float* W2 = (const float*)d_in[3];
  const float* W3 = (const float*)d_in[4];
  const float* W4 = (const float*)d_in[5];
  const float* Wr1 = (const float*)d_in[6];
  const float* Wr2 = (const float*)d_in[7];
  const float* Wd = (const float*)d_in[8];
  const float* bd = (const float*)d_in[9];

  char* ws = (char*)d_ws;
  size_t off = 0;
  auto alloc = [&](size_t bytes) -> char* {
    char* p = ws + off;
    off = (off + bytes + 255) & ~(size_t)255;
    return p;
  };

  _Float16* w1h  = (_Float16*)alloc((size_t)1728 * 2);
  _Float16* w2h  = (_Float16*)alloc((size_t)73728 * 2);
  _Float16* w3h  = (_Float16*)alloc((size_t)294912 * 2);
  _Float16* w4h  = (_Float16*)alloc((size_t)1179648 * 2);
  _Float16* wr1h = (_Float16*)alloc((size_t)1605632 * 2);
  _Float16* wr2h = (_Float16*)alloc((size_t)204800 * 2);
  _Float16* t1 = (_Float16*)alloc((size_t)32 * 128 * 128 * 64 * 2);
  _Float16* t2 = (_Float16*)alloc((size_t)32 * 64 * 64 * 128 * 2);
  _Float16* t3 = (_Float16*)alloc((size_t)32 * 32 * 32 * 256 * 2);
  _Float16* t4 = (_Float16*)alloc((size_t)32 * 16 * 16 * 512 * 2);
  _Float16* fA = (_Float16*)alloc((size_t)32 * 256 * 512 * 2);
  _Float16* fB = (_Float16*)alloc((size_t)32 * 256 * 512 * 2);
  float*    corr   = (float*)alloc((size_t)32 * 65536 * 4);
  _Float16* corr_h = (_Float16*)alloc((size_t)32 * 65536 * 2);
  _Float16* r1 = (_Float16*)alloc((size_t)32 * 100 * 128 * 2);
  _Float16* r2 = (_Float16*)alloc((size_t)32 * 36 * 64 * 2);
  float* geo  = (float*)alloc((size_t)32 * 18 * 4);
  float* tpsb = (float*)alloc((size_t)32 * 40 * 4);

  auto cvt = [&](const float* s, _Float16* d, int K, int CO) {
    const int n = K * CO;
    cvt_transpose<<<(n + 255) / 256, 256, 0, stream>>>(s, d, K, CO);
  };
  cvt(W1, w1h, 27, 64);
  cvt(W2, w2h, 576, 128);
  cvt(W3, w3h, 1152, 256);
  cvt(W4, w4h, 2304, 512);
  cvt(Wr1, wr1h, 12544, 128);
  cvt(Wr2, wr2h, 3200, 64);

  auto feat = [&](const float* img, _Float16* f) {
    conv_relu_wmma<3, 64, 256, 256, 128, 128, 3, 2, float>
        <<<dim3(1024, 4, 32), 32, 0, stream>>>(img, w1h, t1);
    conv_relu_wmma<64, 128, 128, 128, 64, 64, 3, 2, _Float16>
        <<<dim3(256, 8, 32), 32, 0, stream>>>(t1, w2h, t2);
    conv_relu_wmma<128, 256, 64, 64, 32, 32, 3, 2, _Float16>
        <<<dim3(64, 16, 32), 32, 0, stream>>>(t2, w3h, t3);
    conv_relu_wmma<256, 512, 32, 32, 16, 16, 3, 2, _Float16>
        <<<dim3(16, 32, 32), 32, 0, stream>>>(t3, w4h, t4);
    l2norm_512<<<1024, 256, 0, stream>>>(t4, f, 32 * 256);
  };
  feat(imgA, fA);
  feat(imgB, fB);

  corr_wmma<<<dim3(16, 16, 32), 32, 0, stream>>>(fA, fB, corr, corr_h);

  conv_relu_wmma<256, 128, 16, 16, 10, 10, 7, 1, _Float16>
      <<<dim3(7, 8, 32), 32, 0, stream>>>(corr_h, wr1h, r1);
  conv_relu_wmma<128, 64, 10, 10, 6, 6, 5, 1, _Float16>
      <<<dim3(3, 4, 32), 32, 0, stream>>>(r1, wr2h, r2);

  dense_geo<<<32, 32, 0, stream>>>(r2, Wd, bd, geo);
  tps_solve<<<1, 32, 0, stream>>>(geo, tpsb);
  mask_reduce<<<32, 256, 0, stream>>>(corr, tpsb, (float*)d_out);
}
